// GraphormerLayer_62612033241322
// MI455X (gfx1250) — matrix operand order repaired and checked by reading, not compiled, not run on hardware
//
#include <hip/hip_runtime.h>
#include <hip/hip_bf16.h>

// ---------------------------------------------------------------------------
// Graphormer layer for MI455X (gfx1250, wave32, WMMA).
// N=4096, DIM=512, HEADS=8, HEAD_DIM=64.
// Bias stream (512MB) is the bandwidth floor (~22us @ 23.3TB/s); all matmuls
// run through v_wmma_f32_16x16x32_f16 so compute hides under it.  GEMMs use
// CDNA5 async global->LDS copies (ASYNCcnt) with double buffering.
// ---------------------------------------------------------------------------

#define GN    4096
#define GDIM  512
#define GH    8
#define GDH   64
#define GSCALE 0.125f   // 64^-0.5

typedef __attribute__((ext_vector_type(8)))  _Float16 h8;
typedef __attribute__((ext_vector_type(16))) _Float16 v16h;
typedef __attribute__((ext_vector_type(8)))  float    v8f;
typedef __attribute__((ext_vector_type(4)))  int      i32x4;

#if __has_builtin(__builtin_amdgcn_global_load_async_to_lds_b128)
#define HAVE_ASYNC 1
#else
#define HAVE_ASYNC 0
#endif

static __device__ __forceinline__ v16h cat16(h8 lo, h8 hi) {
    return __builtin_shufflevector(lo, hi, 0,1,2,3,4,5,6,7,8,9,10,11,12,13,14,15);
}

static __device__ __forceinline__ v8f wmma_f16(v16h a, v16h b, v8f c) {
    return __builtin_amdgcn_wmma_f32_16x16x32_f16(false, a, false, b, (short)0, c, false, false);
}

// 16B global -> LDS copy: async (no VGPR round trip, ASYNCcnt) when available.
// Builtin signature (from compiler diagnostic): param0 = int4 addrspace(1)*.
static __device__ __forceinline__ void cp16_async(const _Float16* g, _Float16* l) {
#if HAVE_ASYNC
    __builtin_amdgcn_global_load_async_to_lds_b128(
        (__attribute__((address_space(1))) i32x4*)(void*)g,
        (__attribute__((address_space(3))) i32x4*)(void*)l,
        0, 0);
#else
    *(h8*)l = *(const h8*)g;
#endif
}

static __device__ __forceinline__ void wait_async_le3() {
#if HAVE_ASYNC
#if __has_builtin(__builtin_amdgcn_s_wait_asynccnt)
    __builtin_amdgcn_s_wait_asynccnt(3);
#else
    asm volatile("s_wait_asynccnt 0x3" ::: "memory");
#endif
#endif
}
static __device__ __forceinline__ void wait_async_0() {
#if HAVE_ASYNC
#if __has_builtin(__builtin_amdgcn_s_wait_asynccnt)
    __builtin_amdgcn_s_wait_asynccnt(0);
#else
    asm volatile("s_wait_asynccnt 0x0" ::: "memory");
#endif
#endif
}

// XOR-butterfly reductions over the 16-lane half (ds_swizzle, group-of-32:
// imm = (xor_mask<<10) | and_mask(0x1f)); one DS op per step, no fixup VALU.
static __device__ __forceinline__ float half16_max(float v) {
    v = fmaxf(v, __int_as_float(__builtin_amdgcn_ds_swizzle(__float_as_int(v), (1 << 10) | 0x1f)));
    v = fmaxf(v, __int_as_float(__builtin_amdgcn_ds_swizzle(__float_as_int(v), (2 << 10) | 0x1f)));
    v = fmaxf(v, __int_as_float(__builtin_amdgcn_ds_swizzle(__float_as_int(v), (4 << 10) | 0x1f)));
    v = fmaxf(v, __int_as_float(__builtin_amdgcn_ds_swizzle(__float_as_int(v), (8 << 10) | 0x1f)));
    return v;
}
static __device__ __forceinline__ float half16_sum(float v) {
    v += __int_as_float(__builtin_amdgcn_ds_swizzle(__float_as_int(v), (1 << 10) | 0x1f));
    v += __int_as_float(__builtin_amdgcn_ds_swizzle(__float_as_int(v), (2 << 10) | 0x1f));
    v += __int_as_float(__builtin_amdgcn_ds_swizzle(__float_as_int(v), (4 << 10) | 0x1f));
    v += __int_as_float(__builtin_amdgcn_ds_swizzle(__float_as_int(v), (8 << 10) | 0x1f));
    return v;
}

// ---------------------------------------------------------------------------
// fp32 -> f16 elementwise convert
// ---------------------------------------------------------------------------
__global__ __launch_bounds__(256) void cvt_f16_kernel(const float* __restrict__ src,
                                                      _Float16* __restrict__ dst, int n) {
    int i = blockIdx.x * 256 + threadIdx.x;
    if (i < n) dst[i] = (_Float16)src[i];
}

// ---------------------------------------------------------------------------
// W [K x Nout] fp32  ->  Wt [Nout x K] f16   (32x32 LDS tiles)
// ---------------------------------------------------------------------------
__global__ __launch_bounds__(256) void transpose_cvt_kernel(const float* __restrict__ W,
                                                            _Float16* __restrict__ Wt,
                                                            int K, int Nout) {
    __shared__ _Float16 t[32][33];
    int bk = blockIdx.x * 32, bn = blockIdx.y * 32;
    int tx = threadIdx.x & 31, ty = threadIdx.x >> 5;   // 32 x 8
    #pragma unroll
    for (int i = 0; i < 32; i += 8)
        t[ty + i][tx] = (_Float16)W[(size_t)(bk + ty + i) * Nout + bn + tx];
    __syncthreads();
    #pragma unroll
    for (int i = 0; i < 32; i += 8)
        Wt[(size_t)(bn + ty + i) * K + bk + tx] = t[tx][ty + i];
}

// ---------------------------------------------------------------------------
// Generic WMMA GEMM:  C[M x Nc] = A[M x K](f16) * Bt[Nc x K](f16)^T + bias
// Block tile 128x64, 8 waves, each wave 32x32 (2x2 C-tiles).
// K-staging: double-buffered async global->LDS copies (3 insts per tile).
// ---------------------------------------------------------------------------
template<bool RELU, bool HAS_RES, bool OUT_F32, bool OUT_F16>
__global__ __launch_bounds__(256) void gemm_wmma_kernel(
        const _Float16* __restrict__ A, const _Float16* __restrict__ Bt,
        const float* __restrict__ bias, const float* __restrict__ residual,
        float* __restrict__ Cf, _Float16* __restrict__ Ch,
        int M, int Nc, int K) {
    constexpr int BM = 128, BN = 64, BK = 32;
    constexpr int LDA = BK + 8, LDB = BK + 8;      // pads keep 16B alignment (80B rows)
    __shared__ _Float16 As[2][BM * LDA];
    __shared__ _Float16 Bs[2][BN * LDB];

    const int tid  = threadIdx.x;
    const int lane = tid & 31, wave = tid >> 5;
    const int wm0  = (wave & 3) * 32, wn0 = (wave >> 2) * 32;
    const int m0   = blockIdx.x * BM, n0 = blockIdx.y * BN;
    const int h2   = lane >> 4, l16 = lane & 15;

    const int rA = tid >> 1, cA = (tid & 1) * 16;  // A: 2 threads/row, 2x16B
    const int rB = tid >> 2, cB = (tid & 3) * 8;   // B: 4 threads/row, 1x16B

    auto issue = [&](int k0, int p) {
        const _Float16* sA = A + (size_t)(m0 + rA) * K + k0 + cA;
        cp16_async(sA,     &As[p][rA * LDA + cA]);
        cp16_async(sA + 8, &As[p][rA * LDA + cA + 8]);
        cp16_async(Bt + (size_t)(n0 + rB) * K + k0 + cB, &Bs[p][rB * LDB + cB]);
    };

    v8f acc[2][2] = {};
    const int KT = K / BK;
    issue(0, 0);

    for (int kt = 0; kt < KT; ++kt) {
        const int cur = kt & 1;
        if (kt + 1 < KT) { issue((kt + 1) * BK, cur ^ 1); wait_async_le3(); }
        else             { wait_async_0(); }
        __syncthreads();

        v16h af[2], bf[2];
        #pragma unroll
        for (int i = 0; i < 2; ++i) {
            // A fragment 16x32: lane row = l16; half0 K 0-15; half1 K 16-31
            const _Float16* ap = &As[cur][(wm0 + i * 16 + l16) * LDA];
            af[i] = cat16(*(const h8*)(ap + 16 * h2), *(const h8*)(ap + 16 * h2 + 8));
            // B fragment 32x16: lane col = l16; half picks K 0-15 / 16-31 (contig in Wt row)
            const _Float16* bp = &Bs[cur][(wn0 + i * 16 + l16) * LDB];
            bf[i] = cat16(*(const h8*)(bp + 16 * h2), *(const h8*)(bp + 16 * h2 + 8));
        }
        #pragma unroll
        for (int i = 0; i < 2; ++i)
            #pragma unroll
            for (int j = 0; j < 2; ++j)
                acc[i][j] = wmma_f16(af[i], bf[j], acc[i][j]);
        __syncthreads();   // all reads of buf[cur] done before iter kt+1 refills it
    }

    // Epilogue: C/D layout -> row = r + 8*h2, col = l16
    #pragma unroll
    for (int i = 0; i < 2; ++i)
        #pragma unroll
        for (int j = 0; j < 2; ++j) {
            int col = n0 + wn0 + j * 16 + l16;
            float bv = bias ? bias[col] : 0.0f;
            #pragma unroll
            for (int r = 0; r < 8; ++r) {
                int row = m0 + wm0 + i * 16 + r + 8 * h2;
                float v = acc[i][j][r] + bv;
                if (RELU)    v = fmaxf(v, 0.0f);
                if (HAS_RES) v += residual[(size_t)row * Nc + col];
                if (OUT_F32) Cf[(size_t)row * Nc + col] = v;
                if (OUT_F16) Ch[(size_t)row * Nc + col] = (_Float16)v;
            }
        }
}

// ---------------------------------------------------------------------------
// qkv fp32 [N x 1536] -> Qb[h][n][d] f16, Kb[h][n][d] f16, Vt[h][d][n] f16
// ---------------------------------------------------------------------------
__global__ __launch_bounds__(256) void reshape_qkv_kernel(
        const float* __restrict__ qkv,
        _Float16* __restrict__ Qb, _Float16* __restrict__ Kb, _Float16* __restrict__ Vt) {
    int idx = blockIdx.x * 256 + threadIdx.x;      // over N*512
    int n = idx >> 9, c = idx & 511;               // c = h*64 + d
    int h = c >> 6,  d = c & 63;
    const float* p = qkv + (size_t)n * (3 * GDIM);
    _Float16 q = (_Float16)p[c];
    _Float16 k = (_Float16)p[GDIM + c];
    _Float16 v = (_Float16)p[2 * GDIM + c];
    Qb[((size_t)h * GN + n) * GDH + d] = q;
    Kb[((size_t)h * GN + n) * GDH + d] = k;
    Vt[(size_t)c * GN + n]             = v;        // (h*64+d)*N + n
}

// ---------------------------------------------------------------------------
// Flash attention with dense bias.  One wave per (head, 16-query tile).
// Streams keys in blocks of 32; S via 2 chained WMMAs; bias read is THE
// bandwidth floor of the layer (prefetched 2 blocks ahead); P staged through
// wave-private LDS, P*V via 4 WMMAs into running O accumulator.
// ---------------------------------------------------------------------------
__global__ __launch_bounds__(256) void attn_flash_kernel(
        const _Float16* __restrict__ Qb, const _Float16* __restrict__ Kb,
        const _Float16* __restrict__ Vt, const float* __restrict__ bias,
        _Float16* __restrict__ O) {
    constexpr int PST = 40;                        // P row stride (80B, 16B-aligned)
    __shared__ _Float16 Ps[8][16 * PST];

    const int lane = threadIdx.x & 31, wave = threadIdx.x >> 5;
    const int h2 = lane >> 4, l16 = lane & 15;
    const int h  = blockIdx.y;
    const int q0 = (blockIdx.x * 8 + wave) * 16;

    const _Float16* Qh = Qb + ((size_t)h * GN + q0) * GDH;
    const _Float16* Kh = Kb + (size_t)h * GN * GDH;
    const _Float16* Vh = Vt + (size_t)h * GDH * GN;
    const float*    Bh = bias + (size_t)h * GN * GN + (size_t)q0 * GN;
    _Float16* P = &Ps[wave][0];

    // Q fragments (rows q0+l16, dh 0-31 / 32-63)
    v16h qf[2];
    #pragma unroll
    for (int i = 0; i < 2; ++i) {
        const _Float16* qp = Qh + (size_t)l16 * GDH + i * 32;
        qf[i] = cat16(*(const h8*)(qp + 16 * h2), *(const h8*)(qp + 16 * h2 + 8));
    }

    float mrow[8], lrow[8];
    v8f o[4] = {};
    #pragma unroll
    for (int r = 0; r < 8; ++r) { mrow[r] = -3.0e38f; lrow[r] = 0.0f; }

    for (int kb = 0; kb < GN; kb += 32) {
        // prefetch the bias tile 2 key-blocks ahead (lanes cover 16 rows x 2 halves)
        if (kb + 64 < GN)
            __builtin_prefetch(Bh + (size_t)l16 * GN + kb + 64 + 16 * h2, 0, 0);

        // ---- S = Q K^T for 32 keys (two 16x16 f32 tiles) ----
        v8f s[2];
        #pragma unroll
        for (int t = 0; t < 2; ++t) {
            const _Float16* kp = Kh + (size_t)(kb + t * 16 + l16) * GDH;
            v16h b0 = cat16(*(const h8*)(kp + 16 * h2),      *(const h8*)(kp + 16 * h2 + 8));
            v16h b1 = cat16(*(const h8*)(kp + 32 + 16 * h2), *(const h8*)(kp + 32 + 16 * h2 + 8));
            v8f c = {};
            c = wmma_f16(qf[0], b0, c);
            c = wmma_f16(qf[1], b1, c);
            s[t] = c;
        }

        // ---- scale + bias + online softmax (ds_swizzle row reductions) ----
        float alpha[8];
        #pragma unroll
        for (int r = 0; r < 8; ++r) {
            const float* bp = Bh + (size_t)(r + 8 * h2) * GN + kb;
            float v0 = s[0][r] * GSCALE + bp[l16];
            float v1 = s[1][r] * GSCALE + bp[16 + l16];
            float mv = half16_max(fmaxf(v0, v1));
            float mnew = fmaxf(mrow[r], mv);
            alpha[r]   = __expf(mrow[r] - mnew);
            float p0 = __expf(v0 - mnew);
            float p1 = __expf(v1 - mnew);
            float rs = half16_sum(p0 + p1);
            lrow[r] = lrow[r] * alpha[r] + rs;
            mrow[r] = mnew;
            P[(r + 8 * h2) * PST + l16]      = (_Float16)p0;
            P[(r + 8 * h2) * PST + 16 + l16] = (_Float16)p1;
        }
        asm volatile("s_wait_dscnt 0x0" ::: "memory");   // cross-lane LDS handoff (wave-local)

        #pragma unroll
        for (int j = 0; j < 4; ++j)
            #pragma unroll
            for (int r = 0; r < 8; ++r) o[j][r] *= alpha[r];

        // ---- O += P V (A-fragment of P from LDS, B from Vt rows) ----
        const _Float16* pp = P + l16 * PST;
        v16h pf = cat16(*(const h8*)(pp + 16 * h2), *(const h8*)(pp + 16 * h2 + 8));
        #pragma unroll
        for (int j = 0; j < 4; ++j) {
            const _Float16* vp = Vh + (size_t)(j * 16 + l16) * GN + kb + 16 * h2;
            v16h vf = cat16(*(const h8*)(vp), *(const h8*)(vp + 8));
            o[j] = wmma_f16(pf, vf, o[j]);
        }
    }

    // ---- normalize + write [N x 512] at column h*64 + j*16 + l16 ----
    #pragma unroll
    for (int j = 0; j < 4; ++j)
        #pragma unroll
        for (int r = 0; r < 8; ++r) {
            float val = o[j][r] / lrow[r];
            O[(size_t)(q0 + r + 8 * h2) * GDIM + h * GDH + j * 16 + l16] = (_Float16)val;
        }
}

// ---------------------------------------------------------------------------
// LayerNorm over rows of 512; writes fp32 and (optionally) f16.
// ---------------------------------------------------------------------------
template<bool OUT_F16>
__global__ __launch_bounds__(256) void layernorm_kernel(
        const float* __restrict__ X, const float* __restrict__ g, const float* __restrict__ b,
        float* __restrict__ Yf, _Float16* __restrict__ Yh) {
    __shared__ float red[2][8];
    const int row = blockIdx.x, tid = threadIdx.x;
    const int lane = tid & 31, wave = tid >> 5;
    const float* x = X + (size_t)row * GDIM;
    float v0 = x[tid], v1 = x[tid + 256];
    float s = v0 + v1, s2 = v0 * v0 + v1 * v1;
    #pragma unroll
    for (int off = 1; off < 32; off <<= 1) {
        s  += __shfl_xor(s,  off, 32);
        s2 += __shfl_xor(s2, off, 32);
    }
    if (lane == 0) { red[0][wave] = s; red[1][wave] = s2; }
    __syncthreads();
    float ts = 0.0f, ts2 = 0.0f;
    #pragma unroll
    for (int w = 0; w < 8; ++w) { ts += red[0][w]; ts2 += red[1][w]; }
    float mu  = ts * (1.0f / GDIM);
    float var = ts2 * (1.0f / GDIM) - mu * mu;
    float inv = rsqrtf(var + 1e-5f);
    float y0 = (v0 - mu) * inv * g[tid]       + b[tid];
    float y1 = (v1 - mu) * inv * g[tid + 256] + b[tid + 256];
    Yf[(size_t)row * GDIM + tid]       = y0;
    Yf[(size_t)row * GDIM + tid + 256] = y1;
    if (OUT_F16) {
        Yh[(size_t)row * GDIM + tid]       = (_Float16)y0;
        Yh[(size_t)row * GDIM + tid + 256] = (_Float16)y1;
    }
}

// ---------------------------------------------------------------------------
// Host orchestration
// ---------------------------------------------------------------------------
extern "C" void kernel_launch(void* const* d_in, const int* in_sizes, int n_in,
                              void* d_out, int out_size, void* d_ws, size_t ws_size,
                              hipStream_t stream) {
    const float* x       = (const float*)d_in[0];
    const float* abias   = (const float*)d_in[1];
    const float* W_qkv   = (const float*)d_in[2];
    const float* b_qkv   = (const float*)d_in[3];
    const float* W_out   = (const float*)d_in[4];
    const float* b_out   = (const float*)d_in[5];
    const float* W_ffn1  = (const float*)d_in[6];
    const float* b_ffn1  = (const float*)d_in[7];
    const float* W_ffn2  = (const float*)d_in[8];
    const float* b_ffn2  = (const float*)d_in[9];
    const float* ln1_g   = (const float*)d_in[10];
    const float* ln1_b   = (const float*)d_in[11];
    const float* ln2_g   = (const float*)d_in[12];
    const float* ln2_b   = (const float*)d_in[13];
    float* out = (float*)d_out;

    size_t off = 0;
    auto alloc = [&](size_t bytes) -> char* {
        char* p = (char*)d_ws + off;
        off += (bytes + 255) & ~(size_t)255;
        return p;
    };
    _Float16* xb      = (_Float16*)alloc((size_t)GN * GDIM * 2);
    _Float16* WqkvT   = (_Float16*)alloc((size_t)3 * GDIM * GDIM * 2);
    _Float16* WoutT   = (_Float16*)alloc((size_t)GDIM * GDIM * 2);
    _Float16* Wffn1T  = (_Float16*)alloc((size_t)4 * GDIM * GDIM * 2);
    _Float16* Wffn2T  = (_Float16*)alloc((size_t)4 * GDIM * GDIM * 2);
    float*    qkv32   = (float*)   alloc((size_t)GN * 3 * GDIM * 4);
    _Float16* Qb      = (_Float16*)alloc((size_t)GH * GN * GDH * 2);
    _Float16* Kb      = (_Float16*)alloc((size_t)GH * GN * GDH * 2);
    _Float16* Vt      = (_Float16*)alloc((size_t)GH * GDH * GN * 2);
    _Float16* attnO   = (_Float16*)alloc((size_t)GN * GDIM * 2);
    float*    y1      = (float*)   alloc((size_t)GN * GDIM * 4);
    float*    h32     = (float*)   alloc((size_t)GN * GDIM * 4);
    _Float16* h16     = (_Float16*)alloc((size_t)GN * GDIM * 2);
    _Float16* g16     = (_Float16*)alloc((size_t)GN * 4 * GDIM * 2);
    float*    y2      = (float*)   alloc((size_t)GN * GDIM * 4);

    // 1) precision conversion + weight transposes
    cvt_f16_kernel<<<(GN * GDIM) / 256, 256, 0, stream>>>(x, xb, GN * GDIM);
    transpose_cvt_kernel<<<dim3(GDIM / 32, (3 * GDIM) / 32), 256, 0, stream>>>(W_qkv, WqkvT, GDIM, 3 * GDIM);
    transpose_cvt_kernel<<<dim3(GDIM / 32, GDIM / 32),       256, 0, stream>>>(W_out, WoutT, GDIM, GDIM);
    transpose_cvt_kernel<<<dim3(GDIM / 32, (4 * GDIM) / 32), 256, 0, stream>>>(W_ffn1, Wffn1T, GDIM, 4 * GDIM);
    transpose_cvt_kernel<<<dim3((4 * GDIM) / 32, GDIM / 32), 256, 0, stream>>>(W_ffn2, Wffn2T, 4 * GDIM, GDIM);

    // 2) QKV projection (WMMA, async-LDS double buffered)
    gemm_wmma_kernel<false, false, true, false>
        <<<dim3(GN / 128, (3 * GDIM) / 64), 256, 0, stream>>>(
            xb, WqkvT, b_qkv, nullptr, qkv32, nullptr, GN, 3 * GDIM, GDIM);

    // 3) split / transpose heads
    reshape_qkv_kernel<<<(GN * GDIM) / 256, 256, 0, stream>>>(qkv32, Qb, Kb, Vt);

    // 4) flash attention + dense bias (WMMA, streams the 512MB bias once)
    attn_flash_kernel<<<dim3(GN / 128, GH), 256, 0, stream>>>(Qb, Kb, Vt, abias, attnO);

    // 5) out projection + residual(x) -> y1
    gemm_wmma_kernel<false, true, true, false>
        <<<dim3(GN / 128, GDIM / 64), 256, 0, stream>>>(
            attnO, WoutT, b_out, x, y1, nullptr, GN, GDIM, GDIM);

    // 6) LN1 -> h (fp32 + f16)
    layernorm_kernel<true><<<GN, 256, 0, stream>>>(y1, ln1_g, ln1_b, h32, h16);

    // 7) FFN1 + ReLU -> g (f16)
    gemm_wmma_kernel<true, false, false, true>
        <<<dim3(GN / 128, (4 * GDIM) / 64), 256, 0, stream>>>(
            h16, Wffn1T, b_ffn1, nullptr, nullptr, g16, GN, 4 * GDIM, GDIM);

    // 8) FFN2 + residual(h) -> y2
    gemm_wmma_kernel<false, true, true, false>
        <<<dim3(GN / 128, GDIM / 64), 256, 0, stream>>>(
            g16, Wffn2T, b_ffn2, h32, y2, nullptr, GN, GDIM, 4 * GDIM);

    // 9) LN2 -> output (fp32)
    layernorm_kernel<false><<<GN, 256, 0, stream>>>(y2, ln2_g, ln2_b, out, nullptr);
}
